// FPLinearMESO_8194797601027
// MI455X (gfx1250) — 8-bit matrix operand order repaired and checked by reading, compiled, not run on hardware
//
#include <hip/hip_runtime.h>
#include <hip/hip_bf16.h>
#include <stdint.h>

typedef int v8i __attribute__((ext_vector_type(8)));

__global__ void init_slots_kernel(unsigned* slots) {
    slots[0] = 0u;
    slots[1] = 0u;
}

__global__ __launch_bounds__(256)
void absmax_kernel(const float* __restrict__ x, size_t n, unsigned* __restrict__ slot) {
    __shared__ float red[8];
    float m = 0.0f;
    size_t i = ((size_t)blockIdx.x * blockDim.x + threadIdx.x) * 4;
    const size_t stride = (size_t)gridDim.x * blockDim.x * 4;
    for (; i + 3 < n; i += stride) {
        float4 v = *(const float4*)(x + i);
        m = fmaxf(m, fmaxf(fmaxf(fabsf(v.x), fabsf(v.y)),
                           fmaxf(fabsf(v.z), fabsf(v.w))));
    }
    #pragma unroll
    for (int off = 16; off > 0; off >>= 1)
        m = fmaxf(m, __shfl_xor(m, off, 32));
    const int lane = threadIdx.x & 31;
    const int w    = threadIdx.x >> 5;
    if (lane == 0) red[w] = m;
    __syncthreads();
    if (threadIdx.x == 0) {
        float mm = red[0];
        #pragma unroll
        for (int j = 1; j < 8; j++) mm = fmaxf(mm, red[j]);
        atomicMax(slot, __float_as_uint(mm));
    }
}

__global__ __launch_bounds__(256)
void quant_kernel(const float* __restrict__ x, int8_t* __restrict__ q,
                  size_t n, const unsigned* __restrict__ slot) {
    const float am  = __uint_as_float(*slot);
    const float inv = (am == 0.0f) ? 1.0f : (127.0f / am);
    size_t i = ((size_t)blockIdx.x * blockDim.x + threadIdx.x) * 4;
    if (i + 3 >= n) return;
    float4 v = *(const float4*)(x + i);
    int q0 = (int)fminf(fmaxf(rintf(v.x * inv), -128.0f), 127.0f);
    int q1 = (int)fminf(fmaxf(rintf(v.y * inv), -128.0f), 127.0f);
    int q2 = (int)fminf(fmaxf(rintf(v.z * inv), -128.0f), 127.0f);
    int q3 = (int)fminf(fmaxf(rintf(v.w * inv), -128.0f), 127.0f);
    int packed = (q0 & 255) | ((q1 & 255) << 8) | ((q2 & 255) << 16) | ((q3 & 255) << 24);
    *(int*)(q + i) = packed;
}

__global__ __launch_bounds__(128)
void gemm_i8_wmma_kernel(const int8_t* __restrict__ xq,
                         const int8_t* __restrict__ wq,
                         const float* __restrict__ bias,
                         const unsigned* __restrict__ slots,
                         float* __restrict__ out,
                         int M, int N, int K) {
    const int lane = threadIdx.x & 31;
    const int wave = threadIdx.x >> 5;
    const int half = lane >> 4;
    const int l16  = lane & 15;

    const int rowBase = blockIdx.y * 128 + (wave >> 1) * 64;
    const int colBase = blockIdx.x * 128 + (wave & 1) * 64;

    unsigned aOff[4], bOff[4];
    #pragma unroll
    for (int t = 0; t < 4; t++) {
        aOff[t] = (unsigned)(rowBase + t * 16 + l16) * (unsigned)K + (unsigned)(half * 8);
        bOff[t] = (unsigned)(colBase + t * 16 + l16) * (unsigned)K + (unsigned)(half * 8);
    }

    v8i acc[4][4] = {};

    const int8_t* aPtr = xq;
    const int8_t* bPtr = wq;
    for (int k0 = 0; k0 < K; k0 += 64, aPtr += 64, bPtr += 64) {
        v8i a[4], b[4];
        #pragma unroll
        for (int t = 0; t < 4; t++) {
            int2 c0 = *(const int2*)(aPtr + aOff[t]);
            int2 c1 = *(const int2*)(aPtr + aOff[t] + 16);
            int2 c2 = *(const int2*)(aPtr + aOff[t] + 32);
            int2 c3 = *(const int2*)(aPtr + aOff[t] + 48);
            a[t][0] = c0.x; a[t][1] = c0.y;
            a[t][2] = c1.x; a[t][3] = c1.y;
            a[t][4] = c2.x; a[t][5] = c2.y;
            a[t][6] = c3.x; a[t][7] = c3.y;
        }
        #pragma unroll
        for (int t = 0; t < 4; t++) {
            int2 d0 = *(const int2*)(bPtr + bOff[t]);
            int2 d1 = *(const int2*)(bPtr + bOff[t] + 16);
            int2 d2 = *(const int2*)(bPtr + bOff[t] + 32);
            int2 d3 = *(const int2*)(bPtr + bOff[t] + 48);
            b[t][0] = d0.x; b[t][1] = d0.y; b[t][2] = d1.x; b[t][3] = d1.y;
            b[t][4] = d2.x; b[t][5] = d2.y; b[t][6] = d3.x; b[t][7] = d3.y;
        }
        #pragma unroll
        for (int mt = 0; mt < 4; mt++) {
            #pragma unroll
            for (int nt = 0; nt < 4; nt++) {
                acc[mt][nt] = __builtin_amdgcn_wmma_i32_16x16x64_iu8(
                    true, a[mt], true, b[nt], acc[mt][nt], false, false);
            }
        }
    }

    const float ax = __uint_as_float(slots[0]);
    const float aw = __uint_as_float(slots[1]);
    const float sx = (ax == 0.0f) ? 1.0f : (ax / 127.0f);
    const float sw = (aw == 0.0f) ? 1.0f : (aw / 127.0f);
    const float s  = sx * sw;

    #pragma unroll
    for (int nt = 0; nt < 4; nt++) {
        const int col = colBase + nt * 16 + l16;
        const float bv = bias[col];
        #pragma unroll
        for (int mt = 0; mt < 4; mt++) {
            const int rbase = rowBase + mt * 16 + half * 8;
            float* o = out + (size_t)rbase * N + col;
            #pragma unroll
            for (int r = 0; r < 8; r++)
                o[(size_t)r * N] = (float)acc[mt][nt][r] * s + bv;
        }
    }
}

extern "C" void kernel_launch(void* const* d_in, const int* in_sizes, int n_in,
                              void* d_out, int out_size, void* d_ws, size_t ws_size,
                              hipStream_t stream) {
    const float* x    = (const float*)d_in[0];
    const float* w    = (const float*)d_in[1];
    const float* bias = (const float*)d_in[2];
    float* out        = (float*)d_out;

    const int Nout = in_sizes[2];
    const int K    = in_sizes[1] / Nout;
    const int M    = in_sizes[0] / K;

    unsigned* slots = (unsigned*)d_ws;
    int8_t* xq = (int8_t*)d_ws + 256;
    int8_t* wq = xq + (size_t)M * K;

    const size_t nx = (size_t)M * K;
    const size_t nw = (size_t)Nout * K;

    init_slots_kernel<<<1, 1, 0, stream>>>(slots);
    absmax_kernel<<<1024, 256, 0, stream>>>(x, nx, slots + 0);
    absmax_kernel<<<1024, 256, 0, stream>>>(w, nw, slots + 1);
    quant_kernel<<<(unsigned)((nx / 4 + 255) / 256), 256, 0, stream>>>(x, xq, nx, slots + 0);
    quant_kernel<<<(unsigned)((nw / 4 + 255) / 256), 256, 0, stream>>>(w, wq, nw, slots + 1);

    dim3 grid(Nout / 128, M / 128);
    gemm_i8_wmma_kernel<<<grid, 128, 0, stream>>>(xq, wq, bias, slots, out, M, Nout, K);
}
